// DNCCell_37323265802439
// MI455X (gfx1250) — matrix operand order repaired and checked by reading, not compiled, not run on hardware
//
#include <hip/hip_runtime.h>
#include <hip/hip_bf16.h>

#define BATCH 512
#define MEMSZ 32
#define UNITS 1024
#define NHEAD 16
#define DEPTH 8

typedef __attribute__((ext_vector_type(16))) __bf16 bf16x16;
typedef __attribute__((ext_vector_type(8)))  float  f32x8;
typedef __attribute__((ext_vector_type(4)))  unsigned int u32x4_t;
typedef __attribute__((ext_vector_type(8)))  int i32x8_t;
typedef __attribute__((ext_vector_type(4)))  int i32x4_t;

#if __has_builtin(__builtin_amdgcn_tensor_load_to_lds)
#define HAVE_TDM 1
#else
#define HAVE_TDM 0
#endif

__device__ __forceinline__ unsigned short f32_to_bf16(float f) {
    unsigned int u = __float_as_uint(f);
    u += 0x7FFFu + ((u >> 16) & 1u);
    return (unsigned short)(u >> 16);
}

// ---------------------------------------------------------------------------
// Weight transpose + f32->bf16: Wt[l][n][k] = bf16(W[l][k][n]); 32x32 LDS tiles
// ---------------------------------------------------------------------------
__global__ __launch_bounds__(256) void k_transpose_bf16(const float* __restrict__ W,
                                                        unsigned short* __restrict__ Wt) {
    __shared__ float tile[32][33];
    int l  = blockIdx.x >> 10;
    int t  = blockIdx.x & 1023;
    int k0 = (t >> 5) << 5;
    int n0 = (t & 31) << 5;
    size_t base = (size_t)l << 20;
    int cx = threadIdx.x & 31;
    int ry = threadIdx.x >> 5;
#pragma unroll
    for (int i = 0; i < 4; ++i) {
        int r = ry + i * 8;
        tile[r][cx] = W[base + (size_t)(k0 + r) * UNITS + (n0 + cx)];
    }
    __syncthreads();
#pragma unroll
    for (int i = 0; i < 4; ++i) {
        int r = ry + i * 8;
        Wt[base + (size_t)(n0 + r) * UNITS + (k0 + cx)] = f32_to_bf16(tile[cx][r]);
    }
}

// ---------------------------------------------------------------------------
// mem[b][m][h*64+hs] = (state+inputs)[b][(h*32+m)*64+hs]   (initial shuffle)
// ---------------------------------------------------------------------------
__global__ __launch_bounds__(256) void k_init_mem(const float* __restrict__ state,
                                                  const float* __restrict__ inputs,
                                                  float* __restrict__ mem) {
    size_t gid = (size_t)blockIdx.x * 256 + threadIdx.x;
    int b = (int)(gid >> 15);
    int rem = (int)(gid & 32767);
    int m = rem >> 10;
    int u = rem & 1023;
    int h = u >> 6, hs = u & 63;
    size_t src = ((size_t)b << 15) + (size_t)((h * 32 + m) << 6) + hs;
    mem[gid] = state[src] + inputs[src];
}

// ---------------------------------------------------------------------------
// mid-depth shuffle: dst[b][m][h*64+hs] = src[b][c&31][(c>>5)*64+hs], c=m*16+h
// ---------------------------------------------------------------------------
__global__ __launch_bounds__(256) void k_shuffle_mem(const float* __restrict__ src,
                                                     float* __restrict__ dst) {
    size_t gid = (size_t)blockIdx.x * 256 + threadIdx.x;
    int b = (int)(gid >> 15);
    int rem = (int)(gid & 32767);
    int m = rem >> 10;
    int u = rem & 1023;
    int h = u >> 6, hs = u & 63;
    int c = m * 16 + h;
    dst[gid] = src[((size_t)b << 15) + (size_t)((c & 31) << 10) + (size_t)((c >> 5) << 6) + hs];
}

// ---------------------------------------------------------------------------
// Read attention (one block per batch row)
// ---------------------------------------------------------------------------
__global__ __launch_bounds__(256) void k_read_att(const float* __restrict__ mem,
                                                  const float* __restrict__ rW,
                                                  const float* __restrict__ rb,
                                                  float* __restrict__ cmean,
                                                  float* __restrict__ attf,
                                                  unsigned short* __restrict__ attb,
                                                  int save_aux) {
    __shared__ float sm_mean[UNITS];
    __shared__ float sm_log[MEMSZ * NHEAD];
    int b = blockIdx.x, t = threadIdx.x;
    size_t mb = (size_t)b << 15;
#pragma unroll
    for (int i = 0; i < 4; ++i) {
        int u = t + (i << 8);
        float s = 0.f;
        for (int m = 0; m < MEMSZ; ++m) s += mem[mb + (m << 10) + u];
        s *= (1.f / 32.f);
        sm_mean[u] = s;
        if (save_aux) cmean[((size_t)b << 10) + u] = s;
    }
    __syncthreads();
    for (int p = t; p < MEMSZ * NHEAD; p += 256) {
        int m = p >> 4, h = p & 15;
        float s = rb[h];
        const float* mrow = mem + mb + (m << 10);
        for (int u = 0; u < UNITS; ++u)
            s += (mrow[u] + sm_mean[u]) * rW[(u << 4) + h];
        sm_log[p] = s;
    }
    __syncthreads();
    if (t < NHEAD) {
        int h = t;
        float mx = -1e30f;
        for (int m = 0; m < MEMSZ; ++m) mx = fmaxf(mx, sm_log[(m << 4) + h]);
        float sum = 0.f;
        for (int m = 0; m < MEMSZ; ++m) {
            float e = __expf(sm_log[(m << 4) + h] - mx);
            sm_log[(m << 4) + h] = e;
            sum += e;
        }
        float inv = 1.f / sum;
        for (int m = 0; m < MEMSZ; ++m) sm_log[(m << 4) + h] *= inv;
    }
    __syncthreads();
#pragma unroll
    for (int i = 0; i < 4; ++i) {
        int u = t + (i << 8);
        int h = u >> 6;
        float s = 0.f;
        for (int m = 0; m < MEMSZ; ++m) s += sm_log[(m << 4) + h] * mem[mb + (m << 10) + u];
        attf[((size_t)b << 10) + u] = s;
        if (save_aux) attb[((size_t)b << 10) + u] = f32_to_bf16(s);
    }
}

// ---------------------------------------------------------------------------
// TDM: stage a 64-row x 128-K bf16 weight panel into LDS with 16B row padding.
// D# built per CDNA5 ISA ch.8: group0 = {count|flags, lds_addr, gaddr_lo,
// gaddr_hi|type=2}; group1 = dims/tile/strides; groups 2/3 zero (2-D tensor).
// pad_interval=5 (64 DWORDs = one 256B row), pad_amount=3 (4 DWORDs = 16B)
// -> LDS row pitch 272B (16B aligned, bank-staggered).
// ---------------------------------------------------------------------------
#define KC 128                 // K elems per chunk
#define NCHUNK (UNITS / KC)    // 8
#define ROWPITCH 136           // ushorts per LDS row (272 B)
#define SH_HALF (64 * ROWPITCH)      // 8704 ushorts = 17408 B per buffer
#define SH_HALF_BYTES (SH_HALF * 2)

__device__ __forceinline__ void tdm_load_panel(const unsigned short* gsrc, unsigned int lds_off) {
#if HAVE_TDM
    unsigned long long ga = (unsigned long long)(uintptr_t)gsrc;
    u32x4_t g0;
    g0.x = 1u;                                                   // count=1 (valid user D#)
    g0.y = lds_off;                                              // LDS byte address
    g0.z = (unsigned int)(ga & 0xFFFFFFFFu);                     // global_addr[31:0]
    g0.w = (unsigned int)((ga >> 32) & 0x01FFFFFFu) | (2u << 30); // global_addr[56:32] | type=2
    i32x8_t g1;
    g1[0] = (1 << 16) | (1 << 20) | (5 << 22) | (3 << 25);       // data_size=2B, pad_en, int=64DW, amt=4DW
    g1[1] = (int)((UNITS & 0xFFFF) << 16);                       // tensor_dim0 lo16 (=1024)
    g1[2] = (int)(((unsigned)UNITS >> 16) | ((UNITS & 0xFFFF) << 16)); // dim0 hi | tensor_dim1 lo16
    g1[3] = (int)(((unsigned)UNITS >> 16) | (KC << 16));         // dim1 hi | tile_dim0=128
    g1[4] = 64;                                                  // tile_dim1=64, tile_dim2=0
    g1[5] = UNITS;                                               // tensor_dim0_stride lo32 (=1024 elems)
    g1[6] = 0;                                                   // stride hi16 | dim1_stride lo16
    g1[7] = 0;
    i32x4_t z4 = {0, 0, 0, 0};
#if defined(__clang_major__) && __clang_major__ >= 23
    i32x8_t z8 = {0, 0, 0, 0, 0, 0, 0, 0};
    __builtin_amdgcn_tensor_load_to_lds(g0, g1, z4, z4, z8, 0);
#else
    __builtin_amdgcn_tensor_load_to_lds(g0, g1, z4, z4, 0);
#endif
#else
    (void)gsrc; (void)lds_off;
#endif
}

// ---------------------------------------------------------------------------
// bf16 WMMA GEMM: C[512,1024] = A[512,1024] x W[1024,1024] (+bias, opt relu)
// Block = 256 thr (8 waves) -> 128-row x 64-col tile; all waves share one
// TDM-staged, double-buffered LDS weight panel. Each wave: 16x64 strip.
// v_wmma_f32_16x16x32_bf16; A/B k order is contiguous (lane half hi holds k = 16*hi + s).
// ---------------------------------------------------------------------------
template <bool RELU, bool OUT_BF16>
__global__ __launch_bounds__(256) void k_gemm_bf16(const unsigned short* __restrict__ A,
                                                   const unsigned short* __restrict__ Wt,
                                                   const float* __restrict__ bias,
                                                   float* __restrict__ Cf,
                                                   unsigned short* __restrict__ Cb) {
    __shared__ unsigned short sh[2 * SH_HALF];     // assumed at LDS offset 0 (sole shared var)
    const int t    = threadIdx.x;
    const int wid  = t >> 5;
    const int lane = t & 31;
    const int ml   = lane & 15;
    const int hi   = lane >> 4;
    const int bm   = blockIdx.x >> 4;   // 0..3  -> rows bm*128
    const int bn   = blockIdx.x & 15;   // 0..15 -> cols bn*64

    const unsigned short* arow   = A  + (size_t)(bm * 128 + wid * 16 + ml) * UNITS;
    const unsigned short* wpanel = Wt + (size_t)(bn * 64) * UNITS;

    f32x8 c0 = {}, c1 = {}, c2 = {}, c3 = {};

    for (int kc = 0; kc < NCHUNK; ++kc) {
#if HAVE_TDM
        if (wid == 0) {
            if (kc == 0) tdm_load_panel(wpanel, 0);
            __builtin_amdgcn_s_wait_tensorcnt(0);      // chunk kc resident
        }
        __syncthreads();                               // publish buffer; old buffer free
        if (wid == 0 && kc + 1 < NCHUNK)
            tdm_load_panel(wpanel + (size_t)(kc + 1) * KC,
                           (unsigned)(((kc + 1) & 1) * SH_HALF_BYTES));
#else
        __syncthreads();
        // cooperative padded copy fallback: 64 rows x 8 x uint4
        for (int e = t; e < 64 * 8; e += 256) {
            int row = e >> 3, seg = e & 7;
            uint4 v = *(const uint4*)(wpanel + (size_t)row * UNITS + kc * KC + seg * 8);
            *(uint4*)&sh[(kc & 1) * SH_HALF + row * ROWPITCH + seg * 8] = v;
        }
        __syncthreads();
#endif
        const unsigned short* bb = sh + (kc & 1) * SH_HALF;

        // preload A fragments for the whole chunk (4 k-steps)
        union { uint4 q[2]; bf16x16 v; } a[4];
#pragma unroll
        for (int ks = 0; ks < 4; ++ks) {
            const int kk = kc * KC + ks * 32 + hi * 16; // A: lanes<16 K0..15, lanes>=16 K16..31
            a[ks].q[0] = *(const uint4*)(arow + kk);
            a[ks].q[1] = *(const uint4*)(arow + kk + 8);
        }
#pragma unroll
        for (int ks = 0; ks < 4; ++ks) {
            const int kl = ks * 32 + hi * 16;           // B: lanes<16 K0..15, lanes>=16 K16..31
            union { uint4 q[2]; bf16x16 v; } b0, b1, b2, b3;
            const unsigned short* r0 = bb + (0 * 16 + ml) * ROWPITCH + kl;
            const unsigned short* r1 = bb + (1 * 16 + ml) * ROWPITCH + kl;
            const unsigned short* r2 = bb + (2 * 16 + ml) * ROWPITCH + kl;
            const unsigned short* r3 = bb + (3 * 16 + ml) * ROWPITCH + kl;
            b0.q[0] = *(const uint4*)(r0); b0.q[1] = *(const uint4*)(r0 + 8);
            b1.q[0] = *(const uint4*)(r1); b1.q[1] = *(const uint4*)(r1 + 8);
            b2.q[0] = *(const uint4*)(r2); b2.q[1] = *(const uint4*)(r2 + 8);
            b3.q[0] = *(const uint4*)(r3); b3.q[1] = *(const uint4*)(r3 + 8);
            c0 = __builtin_amdgcn_wmma_f32_16x16x32_bf16(false, a[ks].v, false, b0.v, (short)0, c0, false, false);
            c1 = __builtin_amdgcn_wmma_f32_16x16x32_bf16(false, a[ks].v, false, b1.v, (short)0, c1, false, false);
            c2 = __builtin_amdgcn_wmma_f32_16x16x32_bf16(false, a[ks].v, false, b2.v, (short)0, c2, false, false);
            c3 = __builtin_amdgcn_wmma_f32_16x16x32_bf16(false, a[ks].v, false, b3.v, (short)0, c3, false, false);
        }
    }

    const int rowb = bm * 128 + wid * 16 + hi * 8;  // C/D: lane<16 -> M=r, lane>=16 -> M=r+8
#pragma unroll
    for (int r = 0; r < 8; ++r) {
        float vr[4] = {c0[r], c1[r], c2[r], c3[r]};
#pragma unroll
        for (int j = 0; j < 4; ++j) {
            int col = bn * 64 + j * 16 + ml;
            float x = vr[j] + bias[col];
            if (RELU) x = fmaxf(x, 0.0f);
            size_t off = (size_t)(rowb + r) * UNITS + col;
            if (OUT_BF16) Cb[off] = f32_to_bf16(x);
            else          Cf[off] = x;
        }
    }
}

// ---------------------------------------------------------------------------
// LayerNorm over last dim (1024), eps = 1e-3; one block per row
// ---------------------------------------------------------------------------
__global__ __launch_bounds__(256) void k_layernorm(const float* __restrict__ x,
                                                   const float* __restrict__ g,
                                                   const float* __restrict__ be,
                                                   float* __restrict__ y) {
    __shared__ float red[256];
    int b = blockIdx.x, t = threadIdx.x;
    const float* xr = x + ((size_t)b << 10);
    float v0 = xr[t], v1 = xr[t + 256], v2 = xr[t + 512], v3 = xr[t + 768];
    red[t] = v0 + v1 + v2 + v3;
    __syncthreads();
    for (int o = 128; o > 0; o >>= 1) { if (t < o) red[t] += red[t + o]; __syncthreads(); }
    float mean = red[0] * (1.f / 1024.f);
    __syncthreads();
    float d0 = v0 - mean, d1 = v1 - mean, d2 = v2 - mean, d3 = v3 - mean;
    red[t] = d0 * d0 + d1 * d1 + d2 * d2 + d3 * d3;
    __syncthreads();
    for (int o = 128; o > 0; o >>= 1) { if (t < o) red[t] += red[t + o]; __syncthreads(); }
    float rstd = rsqrtf(red[0] * (1.f / 1024.f) + 1e-3f);
    float* yr = y + ((size_t)b << 10);
    yr[t]       = d0 * rstd * g[t]       + be[t];
    yr[t + 256] = d1 * rstd * g[t + 256] + be[t + 256];
    yr[t + 512] = d2 * rstd * g[t + 512] + be[t + 512];
    yr[t + 768] = d3 * rstd * g[t + 768] + be[t + 768];
}

// ---------------------------------------------------------------------------
// Write attention + memory update (one block per batch row)
// ---------------------------------------------------------------------------
__global__ __launch_bounds__(256) void k_write_upd(float* __restrict__ mem,
                                                   const float* __restrict__ cmean,
                                                   const float* __restrict__ v,
                                                   const float* __restrict__ wW,
                                                   const float* __restrict__ wb) {
    __shared__ float sm_key[UNITS];
    __shared__ float sm_log[MEMSZ * NHEAD];
    int b = blockIdx.x, t = threadIdx.x;
    size_t mb = (size_t)b << 15;
#pragma unroll
    for (int i = 0; i < 4; ++i) {
        int u = t + (i << 8);
        sm_key[u] = cmean[((size_t)b << 10) + u] + v[((size_t)b << 10) + u];
    }
    __syncthreads();
    for (int p = t; p < MEMSZ * NHEAD; p += 256) {
        int m = p >> 4, h = p & 15;
        float s = wb[h];
        const float* mrow = mem + mb + (m << 10);
        for (int u = 0; u < UNITS; ++u)
            s += (mrow[u] + sm_key[u]) * wW[(u << 4) + h];
        sm_log[p] = s;
    }
    __syncthreads();
    if (t < NHEAD) {
        int h = t;
        float mx = -1e30f;
        for (int m = 0; m < MEMSZ; ++m) mx = fmaxf(mx, sm_log[(m << 4) + h]);
        float sum = 0.f;
        for (int m = 0; m < MEMSZ; ++m) {
            float e = __expf(sm_log[(m << 4) + h] - mx);
            sm_log[(m << 4) + h] = e;
            sum += e;
        }
        float inv = 1.f / sum;
        for (int m = 0; m < MEMSZ; ++m) sm_log[(m << 4) + h] *= inv;
    }
    __syncthreads();
#pragma unroll
    for (int i = 0; i < 4; ++i) {
        int u = t + (i << 8);
        int h = u >> 6;
        float vv = v[((size_t)b << 10) + u];
        for (int m = 0; m < MEMSZ; ++m) {
            float w_ = sm_log[(m << 4) + h];
            size_t idx = mb + (m << 10) + u;
            mem[idx] = (1.f - w_) * mem[idx] + w_ * vv;
        }
    }
}

// ---------------------------------------------------------------------------
extern "C" void kernel_launch(void* const* d_in, const int* in_sizes, int n_in,
                              void* d_out, int out_size, void* d_ws, size_t ws_size,
                              hipStream_t stream) {
    (void)in_sizes; (void)n_in; (void)out_size; (void)ws_size;
    const float* inputs  = (const float*)d_in[0];
    const float* state   = (const float*)d_in[1];
    const float* read_W  = (const float*)d_in[2];
    const float* read_b  = (const float*)d_in[3];
    const float* write_W = (const float*)d_in[4];
    const float* write_b = (const float*)d_in[5];
    const float* kern_W  = (const float*)d_in[6];
    const float* kern_b  = (const float*)d_in[7];
    const float* proj_W  = (const float*)d_in[8];
    const float* proj_b  = (const float*)d_in[9];
    const float* ln_g    = (const float*)d_in[10];
    const float* ln_b    = (const float*)d_in[11];
    const float* ro_W    = (const float*)d_in[12];
    const float* ro_b    = (const float*)d_in[13];

    float* out_att = (float*)d_out;                       // [512,1024]
    float* out_mem = out_att + (size_t)BATCH * UNITS;     // [512,32,1024] final memory

    char* p = (char*)d_ws;
    float*          memA  = (float*)p;           p += (size_t)BATCH * MEMSZ * UNITS * 4;  // 64 MB
    unsigned short* kWt   = (unsigned short*)p;  p += (size_t)DEPTH * UNITS * UNITS * 2;  // 16 MB
    unsigned short* pWt   = (unsigned short*)p;  p += (size_t)DEPTH * UNITS * UNITS * 2;  // 16 MB
    unsigned short* attb  = (unsigned short*)p;  p += (size_t)BATCH * UNITS * 2;
    unsigned short* v1b   = (unsigned short*)p;  p += (size_t)BATCH * UNITS * 2;
    float*          attf  = (float*)p;           p += (size_t)BATCH * UNITS * 4;
    float*          vpre  = (float*)p;           p += (size_t)BATCH * UNITS * 4;
    float*          vcur  = (float*)p;           p += (size_t)BATCH * UNITS * 4;
    float*          cmean = (float*)p;           p += (size_t)BATCH * UNITS * 4;

    const int elemBlocks = (BATCH * MEMSZ * UNITS) / 256;   // 65536

    k_transpose_bf16<<<DEPTH * 1024, 256, 0, stream>>>(kern_W, kWt);
    k_transpose_bf16<<<DEPTH * 1024, 256, 0, stream>>>(proj_W, pWt);
    k_init_mem<<<elemBlocks, 256, 0, stream>>>(state, inputs, memA);

    float* mem = memA;
    for (int l = 0; l < DEPTH; ++l) {
        if (l == DEPTH / 2) {
            k_shuffle_mem<<<elemBlocks, 256, 0, stream>>>(mem, out_mem);
            mem = out_mem;  // layers 4..7 update memory in place inside d_out
        }
        k_read_att<<<BATCH, 256, 0, stream>>>(mem, read_W + (size_t)l * UNITS * NHEAD,
                                              read_b + (size_t)l * NHEAD,
                                              cmean, attf, attb, 1);
        k_gemm_bf16<true, true><<<64, 256, 0, stream>>>(attb, kWt + (size_t)l * UNITS * UNITS,
                                                        kern_b + (size_t)l * UNITS, nullptr, v1b);
        k_gemm_bf16<false, false><<<64, 256, 0, stream>>>(v1b, pWt + (size_t)l * UNITS * UNITS,
                                                          proj_b + (size_t)l * UNITS, vpre, nullptr);
        k_layernorm<<<BATCH, 256, 0, stream>>>(vpre, ln_g, ln_b, vcur);
        k_write_upd<<<BATCH, 256, 0, stream>>>(mem, cmean, vcur,
                                               write_W + (size_t)l * UNITS * NHEAD,
                                               write_b + (size_t)l * NHEAD);
    }
    // final readout; memory already resides in d_out's mem region
    k_read_att<<<BATCH, 256, 0, stream>>>(mem, ro_W, ro_b, cmean, out_att, attb, 0);
}
